// GPT_63419487093418
// MI455X (gfx1250) — matrix operand order repaired and checked by reading, not compiled, not run on hardware
//
#include <hip/hip_runtime.h>
#include <hip/hip_bf16.h>

// ---------------- model constants ----------------
static constexpr int Lc = 6, Bc = 4, Tc = 512, Cc = 384, Hc = 6, Ec = 8, Vc = 50304;
static constexpr int Dh = Cc / Hc;          // 64
static constexpr int Fc = 4 * Cc;           // 1536
static constexpr int Nc = Bc * Tc;          // 2048
static constexpr int CAP = 512;             // N*K/E
static constexpr float LN_EPS = 1e-5f;

typedef __attribute__((ext_vector_type(16))) __bf16 v16bf;
typedef __attribute__((ext_vector_type(4)))  __bf16 v4bf;
typedef __attribute__((ext_vector_type(8)))  float  v8f;
typedef __attribute__((ext_vector_type(4)))  float  v4f;

enum { EPI_STORE = 0, EPI_RELU = 1, EPI_ACCUM = 2, EPI_SCATTER = 3 };

// ---- global-address-space helpers: force global_load/store, not flat ----
__device__ __forceinline__ v4f gload4(const float* p) {
    return *(const __attribute__((address_space(1))) v4f*)p;
}
__device__ __forceinline__ float gload1(const float* p) {
    return *(const __attribute__((address_space(1))) float*)p;
}
__device__ __forceinline__ void gstore1(float* p, float v) {
    *(__attribute__((address_space(1))) float*)p = v;
}
// native bf16 convert (fptrunc RNE) + packed 8B LDS store
__device__ __forceinline__ void stbf4(__bf16* p, v4f v) {
    v4bf o; o.x = (__bf16)v.x; o.y = (__bf16)v.y; o.z = (__bf16)v.z; o.w = (__bf16)v.w;
    *(v4bf*)p = o;
}

union FragBF { v16bf v; __bf16 e[16]; };

// ---------------- specialized bf16-WMMA GEMM ----------------
// Static preconditions (hold for every launch below): M % 64 == 0, K % 32 == 0,
// N % 4 == 0, all leading dims multiples of 4 floats (16B-aligned rows).
// NGUARD: only needed when Nn % 128 != 0 (sole case: the p@v GEMM, Nn = 64).
template<bool TRANSB, bool HAS_IDX, int EPI, bool NGUARD>
__launch_bounds__(256)
__global__ void gemm_bf16_wmma(
    const float* __restrict__ A,  long long bsA, int lda,
    const float* __restrict__ Bm, long long bsB, int ldb,
    float* __restrict__ Cd,       long long bsC, int ldc,
    const float* __restrict__ bias, int biasStride,
    const int*  __restrict__ rowIdxA,
    const int*  __restrict__ outIdx, const float* __restrict__ rowW,
    int idxStride,
    int M, int Nn, int Kk)
{
    (void)M;
    constexpr int BM = 64, BN = 128, BK = 32, PAD = 36; // 72B pitch: 8B-aligned, conflict-free
    __shared__ __bf16 As[BM][PAD];
    __shared__ __bf16 Bs[BN][PAD];

    const int t    = threadIdx.x;
    const int z    = blockIdx.z;
    const int m0   = blockIdx.y * BM;
    const int n0   = blockIdx.x * BN;
    const int lane = t & 31;
    const int wave = t >> 5;
    const int wm   = wave & 1;       // 2 wave rows
    const int wn   = wave >> 1;      // 4 wave cols
    const int half = lane >> 4;
    const int lr   = lane & 15;

    v8f acc[2][2];
    v8f vzero = {0.f,0.f,0.f,0.f,0.f,0.f,0.f,0.f};
    acc[0][0] = vzero; acc[0][1] = vzero; acc[1][0] = vzero; acc[1][1] = vzero;
    const v4f zero4 = {0.f, 0.f, 0.f, 0.f};

    const float* Az = A  + (long long)z * bsA;
    const float* Bz = Bm + (long long)z * bsB;

    // ---- A staging state: 2 float4 chunks / thread, row pointers hoisted ----
    const int arow0 = t >> 3;          // rows 0..31, second chunk row+32
    const int akoff = (t & 7) * 4;     // 16B-aligned k offset
    const float* ap0;
    const float* ap1;
    {
        int gm0 = m0 + arow0, gm1 = gm0 + 32;
        long long r0 = HAS_IDX ? (long long)rowIdxA[(long long)z * idxStride + gm0] : (long long)gm0;
        long long r1 = HAS_IDX ? (long long)rowIdxA[(long long)z * idxStride + gm1] : (long long)gm1;
        ap0 = Az + r0 * (long long)lda + akoff;
        ap1 = Az + r1 * (long long)lda + akoff;
    }

    // ---- B staging state ----
    const int bn4 = (t & 31) * 4;      // !TRANSB: 4-col group base
    const int bk0 = t >> 5;            // !TRANSB: k sub-row 0..7
    const bool cOK = !NGUARD || (n0 + bn4) < Nn;  // Nn % 4 == 0
    const float* bcol = Bz + (n0 + bn4);
    const int ttn  = t >> 3;           // TRANSB: base row 0..31
    const int tk4  = (t & 7) * 4;      // TRANSB: 16B-aligned k offset

    for (int k0 = 0; k0 < Kk; k0 += BK) {
        __builtin_prefetch(ap0 + BK, 0, 1);
        // --- stage A tile (64x32): 2x global_load_b128 + 2x ds_store_b64 ---
        v4f va0 = gload4(ap0); ap0 += BK;
        v4f va1 = gload4(ap1); ap1 += BK;
        stbf4(&As[arow0][akoff], va0);
        stbf4(&As[arow0 + 32][akoff], va1);
        // --- stage B tile as [n][k] (B^T) ---
        if (TRANSB) { // B stored N x K row-major: contiguous in k, direct copy
            #pragma unroll
            for (int j = 0; j < 4; ++j) {
                int n = j * 32 + ttn;
                int gn = n0 + n;
                v4f vb = (!NGUARD || gn < Nn)
                             ? gload4(Bz + (long long)gn * ldb + k0 + tk4) : zero4;
                stbf4(&Bs[n][tk4], vb);
            }
        } else {      // B stored K x N row-major: coalesced b128 in n, transpose into LDS
            #pragma unroll
            for (int j = 0; j < 4; ++j) {
                int kk = j * 8 + bk0;
                v4f vb = cOK ? gload4(bcol + (long long)(k0 + kk) * ldb) : zero4;
                Bs[bn4 + 0][kk] = (__bf16)vb.x;
                Bs[bn4 + 1][kk] = (__bf16)vb.y;
                Bs[bn4 + 2][kk] = (__bf16)vb.z;
                Bs[bn4 + 3][kk] = (__bf16)vb.w;
            }
        }
        __syncthreads();

        // --- build fragments per ISA VGPR layouts, 4 WMMAs per wave ---
        FragBF a[2], b[2];
        #pragma unroll
        for (int ti = 0; ti < 2; ++ti) {
            int m = wm * 32 + ti * 16 + lr;               // A: lane -> row M
            #pragma unroll
            for (int e2 = 0; e2 < 8; ++e2) {              // K = g*16 + half*8 + e
                a[ti].e[e2]     = As[m][half * 8 + e2];
                a[ti].e[8 + e2] = As[m][16 + half * 8 + e2];
            }
        }
        #pragma unroll
        for (int tj = 0; tj < 2; ++tj) {
            int n = wn * 32 + tj * 16 + lr;               // B: lane -> col N
            #pragma unroll
            for (int e2 = 0; e2 < 8; ++e2) {              // K = g*16 + half*8 + e
                b[tj].e[e2]     = Bs[n][half * 8 + e2];
                b[tj].e[8 + e2] = Bs[n][16 + half * 8 + e2];
            }
        }
        #pragma unroll
        for (int ti = 0; ti < 2; ++ti)
            #pragma unroll
            for (int tj = 0; tj < 2; ++tj)
                acc[ti][tj] = __builtin_amdgcn_wmma_f32_16x16x32_bf16(
                    false, a[ti].v, false, b[tj].v, (short)0, acc[ti][tj], false, false);
        __syncthreads();
    }

    // --- epilogue: D layout M = j + 8*half, N = lr; rows always in-bounds ---
    float* Cz = Cd + (long long)z * bsC;
    #pragma unroll
    for (int ti = 0; ti < 2; ++ti) {
        #pragma unroll
        for (int tj = 0; tj < 2; ++tj) {
            int n = n0 + wn * 32 + tj * 16 + lr;
            if (NGUARD && n >= Nn) continue;
            float bv = bias ? gload1(bias + (long long)z * biasStride + n) : 0.f;
            #pragma unroll
            for (int j = 0; j < 8; ++j) {
                int m = m0 + wm * 32 + ti * 16 + half * 8 + j;
                float val = acc[ti][tj][j] + bv;
                if (EPI == EPI_RELU)   val = fmaxf(val, 0.f);
                if (EPI == EPI_SCATTER) {
                    long long orow = outIdx[(long long)z * idxStride + m];
                    float wgt = rowW[(long long)z * idxStride + m];
                    atomicAdd(&Cz[orow * (long long)ldc + n], val * wgt);
                } else if (EPI == EPI_ACCUM) {
                    float* p = &Cz[(long long)m * ldc + n];
                    gstore1(p, gload1(p) + val);
                } else {
                    gstore1(&Cz[(long long)m * ldc + n], val);
                }
            }
        }
    }
}

// ---------------- embedding: one block per token row ----------------
__global__ void embed_kernel(const int* __restrict__ idx, const float* __restrict__ wte,
                             const float* __restrict__ wpe, float* __restrict__ x)
{
    int n = blockIdx.x, c = threadIdx.x;         // blockDim.x == Cc
    int tpos = n & (Tc - 1);
    gstore1(&x[(long long)n * Cc + c],
            gload1(&wte[(long long)idx[n] * Cc + c]) + gload1(&wpe[(long long)tpos * Cc + c]));
}

// ---------------- layernorm (1 block / row, 128 threads) ----------------
__global__ void layernorm_kernel(const float* __restrict__ in, float* __restrict__ out,
                                 const float* __restrict__ g, const float* __restrict__ b)
{
    __shared__ float buf[128];
    int n = blockIdx.x, t = threadIdx.x;
    const float* r = in + (long long)n * Cc;
    float s = 0.f;
    for (int c = t; c < Cc; c += 128) s += gload1(r + c);
    buf[t] = s; __syncthreads();
    for (int st = 64; st > 0; st >>= 1) { if (t < st) buf[t] += buf[t + st]; __syncthreads(); }
    float mu = buf[0] / Cc; __syncthreads();
    float v = 0.f;
    for (int c = t; c < Cc; c += 128) { float d = gload1(r + c) - mu; v += d * d; }
    buf[t] = v; __syncthreads();
    for (int st = 64; st > 0; st >>= 1) { if (t < st) buf[t] += buf[t + st]; __syncthreads(); }
    float rs = rsqrtf(buf[0] / Cc + LN_EPS);
    for (int c = t; c < Cc; c += 128)
        gstore1(&out[(long long)n * Cc + c], (gload1(r + c) - mu) * rs * g[c] + b[c]);
}

// ---------------- RoPE + head split ----------------
__global__ void rope_split_kernel(const float* __restrict__ qkv, float* __restrict__ q,
                                  float* __restrict__ k, float* __restrict__ v)
{
    constexpr int D2 = Dh / 2;
    int i = blockIdx.x * 256 + threadIdx.x;
    if (i >= Bc * Hc * Tc * D2) return;
    int d = i & (D2 - 1);
    int tpos = (i >> 5) & (Tc - 1);
    int h = (i >> 14) % Hc;
    int b = i / (D2 * Tc * Hc);
    int n = b * Tc + tpos;
    const float* row = qkv + (long long)n * 3 * Cc;
    float fr = (float)tpos * powf(10000.f, -2.f * (float)d / (float)Dh);
    float cs = cosf(fr), sn = sinf(fr);
    int c1 = h * Dh + d, c2 = c1 + D2;
    long long ob = (((long long)b * Hc + h) * Tc + tpos) * Dh;
    float q1 = gload1(row + c1), q2 = gload1(row + c2);
    gstore1(&q[ob + d],       q1 * cs + q2 * sn);
    gstore1(&q[ob + d + D2], -q1 * sn + q2 * cs);
    float k1 = gload1(row + Cc + c1), k2 = gload1(row + Cc + c2);
    gstore1(&k[ob + d],       k1 * cs + k2 * sn);
    gstore1(&k[ob + d + D2], -k1 * sn + k2 * cs);
    gstore1(&v[ob + d],      gload1(row + 2 * Cc + c1));
    gstore1(&v[ob + d + D2], gload1(row + 2 * Cc + c2));
}

// ---------------- causal softmax (scale fused, exp cached in-place) ----------------
__global__ void softmax_causal_kernel(float* __restrict__ scores)
{
    __shared__ float buf[256];
    long long row = blockIdx.x;           // over B*H*T rows
    int qpos = (int)(row & (Tc - 1));
    float* p = scores + row * Tc;
    int t = threadIdx.x;
    const float scale = 0.125f;           // 1/sqrt(64)
    float mx = -3.4e38f;
    for (int j = t; j <= qpos; j += 256) mx = fmaxf(mx, gload1(p + j) * scale);
    buf[t] = mx; __syncthreads();
    for (int st = 128; st > 0; st >>= 1) { if (t < st) buf[t] = fmaxf(buf[t], buf[t + st]); __syncthreads(); }
    mx = buf[0]; __syncthreads();
    float s = 0.f;
    for (int j = t; j <= qpos; j += 256) {
        float e = expf(gload1(p + j) * scale - mx);
        gstore1(p + j, e);                // cache exp
        s += e;
    }
    buf[t] = s; __syncthreads();
    for (int st = 128; st > 0; st >>= 1) { if (t < st) buf[t] += buf[t + st]; __syncthreads(); }
    float inv = 1.f / buf[0];
    for (int j = t; j < Tc; j += 256)
        gstore1(p + j, (j <= qpos) ? gload1(p + j) * inv : 0.f);
}

// ---------------- merge heads (B,H,T,D) -> (N,C): one block per token ----------------
__global__ void merge_heads_kernel(const float* __restrict__ oh, float* __restrict__ of)
{
    int n = blockIdx.x, c = threadIdx.x;          // blockDim.x == Cc
    int h = c >> 6, d = c & (Dh - 1);
    int b = n >> 9, tpos = n & (Tc - 1);
    gstore1(&of[(long long)n * Cc + c],
            gload1(&oh[(((long long)b * Hc + h) * Tc + tpos) * Dh + d]));
}

// ---------------- noisy top-2 router ----------------
__global__ void router_kernel(const float* __restrict__ h,
                              const float* __restrict__ Wr, const float* __restrict__ br,
                              const float* __restrict__ Wn, const float* __restrict__ bn,
                              const float* __restrict__ noise,
                              int* __restrict__ topi, float* __restrict__ gw)
{
    __shared__ float sh[32][Ec];
    int t = threadIdx.x;
    int tl = t >> 3, e = t & 7;
    int n = blockIdx.x * 32 + tl;
    if (n < Nc) {
        const float* hr = h + (long long)n * Cc;
        float aR = br[e], aN = bn[e];
        for (int c = 0; c < Cc; ++c) {
            float hv = gload1(hr + c);
            aR += hv * gload1(Wr + (long long)c * Ec + e);
            aN += hv * gload1(Wn + (long long)c * Ec + e);
        }
        float sp = (aN > 20.f) ? aN : log1pf(expf(aN));   // softplus
        sh[tl][e] = aR + gload1(noise + (long long)n * Ec + e) * sp;
    }
    __syncthreads();
    if (n < Nc && e == 0) {
        int i1 = 0, i2 = -1; float v1 = -3.4e38f, v2 = -3.4e38f;
        for (int j = 0; j < Ec; ++j) {
            float vv = sh[tl][j];
            if (vv > v1)      { v2 = v1; i2 = i1; v1 = vv; i1 = j; }
            else if (vv > v2) { v2 = vv; i2 = j; }
        }
        float e1 = 1.f, e2 = expf(v2 - v1), s = e1 + e2;
        topi[2 * n]     = i1; topi[2 * n + 1] = i2;
        gw[2 * n]       = e1 / s; gw[2 * n + 1] = e2 / s;
    }
}

// ---------------- capacity-limited expert lists (stable token order) ----------------
__global__ void build_lists_kernel(const int* __restrict__ topi, const float* __restrict__ gw,
                                   int* __restrict__ order, float* __restrict__ wexp)
{
    int e = threadIdx.x;
    if (e >= Ec) return;
    int cnt = 0;
    for (int n = 0; n < Nc && cnt < CAP; ++n) {
        float g = -1.f;
        if (topi[2 * n] == e)          g = gw[2 * n];
        else if (topi[2 * n + 1] == e) g = gw[2 * n + 1];
        if (g >= 0.f) { order[e * CAP + cnt] = n; wexp[e * CAP + cnt] = g; ++cnt; }
    }
    for (; cnt < CAP; ++cnt) { order[e * CAP + cnt] = 0; wexp[e * CAP + cnt] = 0.f; }
}

// ---------------- host-side orchestration ----------------
extern "C" void kernel_launch(void* const* d_in, const int* in_sizes, int n_in,
                              void* d_out, int out_size, void* d_ws, size_t ws_size,
                              hipStream_t stream)
{
    (void)in_sizes; (void)n_in; (void)out_size; (void)ws_size;

    const int*   idx   = (const int*)  d_in[0];
    const float* noise = (const float*)d_in[1];
    const float* wte   = (const float*)d_in[2];
    const float* wpe   = (const float*)d_in[3];
    const float* ln1_g = (const float*)d_in[4];
    const float* ln1_b = (const float*)d_in[5];
    const float* Wqkv  = (const float*)d_in[6];
    const float* bqkv  = (const float*)d_in[7];
    const float* Wproj = (const float*)d_in[8];
    const float* bproj = (const float*)d_in[9];
    const float* ln2_g = (const float*)d_in[10];
    const float* ln2_b = (const float*)d_in[11];
    const float* Wr    = (const float*)d_in[12];
    const float* br    = (const float*)d_in[13];
    const float* Wn    = (const float*)d_in[14];
    const float* bn    = (const float*)d_in[15];
    const float* We1   = (const float*)d_in[16];
    const float* be1   = (const float*)d_in[17];
    const float* We2   = (const float*)d_in[18];
    const float* be2   = (const float*)d_in[19];
    const float* lnf_g = (const float*)d_in[20];
    const float* lnf_b = (const float*)d_in[21];
    float* out = (float*)d_out;

    // workspace bump allocator (256B aligned)
    char* wsb = (char*)d_ws;
    size_t off = 0;
    auto alloc = [&](size_t bytes) -> void* {
        void* p = wsb + off;
        off += (bytes + 255) & ~(size_t)255;
        return p;
    };
    float* x     = (float*)alloc((size_t)Nc * Cc * 4);
    float* h     = (float*)alloc((size_t)Nc * Cc * 4);
    float* qkv   = (float*)alloc((size_t)Nc * 3 * Cc * 4);
    float* qb    = (float*)alloc((size_t)Nc * Cc * 4);
    float* kb    = (float*)alloc((size_t)Nc * Cc * 4);
    float* vb    = (float*)alloc((size_t)Nc * Cc * 4);
    float* sc    = (float*)alloc((size_t)Bc * Hc * Tc * Tc * 4);
    float* oh    = (float*)alloc((size_t)Nc * Cc * 4);
    float* of    = (float*)alloc((size_t)Nc * Cc * 4);
    float* H1    = (float*)alloc((size_t)Ec * CAP * Fc * 4);
    int*   topi  = (int*)  alloc((size_t)Nc * 2 * 4);
    float* gwv   = (float*)alloc((size_t)Nc * 2 * 4);
    int*   order = (int*)  alloc((size_t)Ec * CAP * 4);
    float* wexp  = (float*)alloc((size_t)Ec * CAP * 4);

    auto cdiv = [](int a, int b) { return (a + b - 1) / b; };

    // x = wte[idx] + wpe
    embed_kernel<<<Nc, Cc, 0, stream>>>(idx, wte, wpe, x);

    for (int l = 0; l < Lc; ++l) {
        // ---- attention ----
        layernorm_kernel<<<Nc, 128, 0, stream>>>(x, h, ln1_g + l * Cc, ln1_b + l * Cc);

        gemm_bf16_wmma<false,false,EPI_STORE,false>
            <<<dim3(cdiv(3 * Cc, 128), cdiv(Nc, 64), 1), 256, 0, stream>>>(
            h, 0, Cc, Wqkv + (long long)l * Cc * 3 * Cc, 0, 3 * Cc,
            qkv, 0, 3 * Cc, bqkv + (long long)l * 3 * Cc, 0,
            nullptr, nullptr, nullptr, 0, Nc, 3 * Cc, Cc);

        rope_split_kernel<<<cdiv(Bc * Hc * Tc * (Dh / 2), 256), 256, 0, stream>>>(qkv, qb, kb, vb);

        // scores = q @ k^T  (batched over B*H, transB)
        gemm_bf16_wmma<true,false,EPI_STORE,false>
            <<<dim3(cdiv(Tc, 128), cdiv(Tc, 64), Bc * Hc), 256, 0, stream>>>(
            qb, (long long)Tc * Dh, Dh, kb, (long long)Tc * Dh, Dh,
            sc, (long long)Tc * Tc, Tc, nullptr, 0,
            nullptr, nullptr, nullptr, 0, Tc, Tc, Dh);

        softmax_causal_kernel<<<Bc * Hc * Tc, 256, 0, stream>>>(sc);

        // o = p @ v  (batched over B*H; Nn = 64 -> needs N guard)
        gemm_bf16_wmma<false,false,EPI_STORE,true>
            <<<dim3(cdiv(Dh, 128), cdiv(Tc, 64), Bc * Hc), 256, 0, stream>>>(
            sc, (long long)Tc * Tc, Tc, vb, (long long)Tc * Dh, Dh,
            oh, (long long)Tc * Dh, Dh, nullptr, 0,
            nullptr, nullptr, nullptr, 0, Tc, Dh, Tc);

        merge_heads_kernel<<<Nc, Cc, 0, stream>>>(oh, of);

        // x += o @ Wproj + bproj   (fused residual accumulate)
        gemm_bf16_wmma<false,false,EPI_ACCUM,false>
            <<<dim3(cdiv(Cc, 128), cdiv(Nc, 64), 1), 256, 0, stream>>>(
            of, 0, Cc, Wproj + (long long)l * Cc * Cc, 0, Cc,
            x, 0, Cc, bproj + (long long)l * Cc, 0,
            nullptr, nullptr, nullptr, 0, Nc, Cc, Cc);

        // ---- MoE ----
        layernorm_kernel<<<Nc, 128, 0, stream>>>(x, h, ln2_g + l * Cc, ln2_b + l * Cc);

        router_kernel<<<cdiv(Nc, 32), 256, 0, stream>>>(
            h, Wr + (long long)l * Cc * Ec, br + (long long)l * Ec,
            Wn + (long long)l * Cc * Ec, bn + (long long)l * Ec,
            noise + (long long)l * Nc * Ec, topi, gwv);

        build_lists_kernel<<<1, 32, 0, stream>>>(topi, gwv, order, wexp);

        // H1[e] = relu(gather(h, order[e]) @ We1[e] + be1[e])   batched over experts
        gemm_bf16_wmma<false,true,EPI_RELU,false>
            <<<dim3(cdiv(Fc, 128), cdiv(CAP, 64), Ec), 256, 0, stream>>>(
            h, 0, Cc, We1 + (long long)l * Ec * Cc * Fc, (long long)Cc * Fc, Fc,
            H1, (long long)CAP * Fc, Fc, be1 + (long long)l * Ec * Fc, Fc,
            order, nullptr, nullptr, CAP, CAP, Fc, Cc);

        // x[token] += gate * (H1[e] @ We2[e] + be2[e])   weighted atomic scatter
        gemm_bf16_wmma<false,false,EPI_SCATTER,false>
            <<<dim3(cdiv(Cc, 128), cdiv(CAP, 64), Ec), 256, 0, stream>>>(
            H1, (long long)CAP * Fc, Fc, We2 + (long long)l * Ec * Fc * Cc, (long long)Fc * Cc, Cc,
            x, 0, Cc, be2 + (long long)l * Ec * Cc, Cc,
            nullptr, order, wexp, CAP, CAP, Cc, Fc);
    }

    // final layernorm + logits = h @ wte^T
    layernorm_kernel<<<Nc, 128, 0, stream>>>(x, h, lnf_g, lnf_b);

    gemm_bf16_wmma<true,false,EPI_STORE,false>
        <<<dim3(cdiv(Vc, 128), cdiv(Nc, 64), 1), 256, 0, stream>>>(
        h, 0, Cc, wte, 0, Cc,
        out, 0, Vc, nullptr, 0,
        nullptr, nullptr, nullptr, 0, Nc, Vc, Cc);
}
